// MessagePassing_77421080478011
// MI455X (gfx1250) — matrix operand order repaired and checked by reading, not compiled, not run on hardware
//
#include <hip/hip_runtime.h>

// ---------------------------------------------------------------------------
// Fused message-passing network for MI455X (gfx1250, wave32, WMMA).
// D = 128, M = 8 modules, 2 steps, batch 65536.
// f16 WMMA 16x16x32 for all four GEMMs; f32 residual state in LDS.
// Round-3 change: keep K loops rolled (#pragma unroll 1) so loop-invariant
// weight (B) fragments cannot be hoisted+spilled to scratch; only one
// K-step of B fragments (32 VGPRs) is live at a time.
// ---------------------------------------------------------------------------

#define D        128
#define NMOD     8
#define BPB      8          // batch elements per block
#define ROWS     64         // BPB * NMOD rows per block
#define THREADS  256        // 8 waves: wave>>1 = M-tile (16 rows), wave&1 = N-half (64 cols)
#define NSTEPS   2
#define BATCH    65536

typedef __attribute__((ext_vector_type(16))) _Float16 v16h;
typedef __attribute__((ext_vector_type(8)))  float    v8f;

union FragU { v16h v; unsigned int u[8]; };

__device__ __forceinline__ unsigned int pack2h(float a, float b) {
    union { _Float16 h[2]; unsigned int u; } x;
    x.h[0] = (_Float16)a; x.h[1] = (_Float16)b;
    return x.u;
}

// A fragment (16x32, f16) from an f16 LDS tile, row-major stride D.
// Layout (ISA 7.12.2): lane&15 = M row, lane>>4 selects K octet;
// VGPR v<4 -> K = 8*half + 2v, v>=4 -> K = 16 + 8*half + 2(v-4).
__device__ __forceinline__ v16h loadA_h(const _Float16* A, int mBase, int kBase, int lane) {
    FragU f;
    const int m    = mBase + (lane & 15);
    const int half = (lane >> 4) & 1;
    const unsigned int* p0 = (const unsigned int*)(A + m * D + kBase + 8 * half);
    const unsigned int* p1 = (const unsigned int*)(A + m * D + kBase + 16 + 8 * half);
    f.u[0] = p0[0]; f.u[1] = p0[1]; f.u[2] = p0[2]; f.u[3] = p0[3];
    f.u[4] = p1[0]; f.u[5] = p1[1]; f.u[6] = p1[2]; f.u[7] = p1[3];
    return f.v;
}

// A fragment from an f32 LDS tile, converting f32 -> f16 on the fly.
__device__ __forceinline__ v16h loadA_f(const float* A, int mBase, int kBase, int lane) {
    FragU f;
    const int m    = mBase + (lane & 15);
    const int half = (lane >> 4) & 1;
    const float* p0 = A + m * D + kBase + 8 * half;
    const float* p1 = p0 + 16;
    f.u[0] = pack2h(p0[0], p0[1]); f.u[1] = pack2h(p0[2], p0[3]);
    f.u[2] = pack2h(p0[4], p0[5]); f.u[3] = pack2h(p0[6], p0[7]);
    f.u[4] = pack2h(p1[0], p1[1]); f.u[5] = pack2h(p1[2], p1[3]);
    f.u[6] = pack2h(p1[4], p1[5]); f.u[7] = pack2h(p1[6], p1[7]);
    return f.v;
}

// B fragment (32x16, f16) from global f16 weights W[K][128] (K-major).
// Layout: lane&15 = N column; element s -> K = 8*half + s (s<8), 16 + 8*half + (s-8) (s>=8).
__device__ __forceinline__ v16h loadB_h(const _Float16* __restrict__ W,
                                        int kBase, int nBase, int lane) {
    FragU f;
    const int n    = nBase + (lane & 15);
    const int half = (lane >> 4) & 1;
#pragma unroll
    for (int s = 0; s < 16; ++s) {
        const int k = kBase + (s < 8 ? 8 * half + s : 16 + 8 * half + (s - 8));
        f.v[s] = W[(size_t)k * D + n];
    }
    return f.v;
}

#define WMMA_F16(acc, a, b)                                                   \
    (acc) = __builtin_amdgcn_wmma_f32_16x16x32_f16(false, (a), false, (b),    \
                                                   (short)0, (acc), false, false)

// Accumulate a 16x64 slab: 4 N-tiles, KSTEPS K-steps of 32.
// First KSWITCH K-steps read A from the f32 state tile (converted on the
// fly); the rest read A from the f16 tile Ah (offset rebased).
// K loops are intentionally NOT unrolled: keeps exactly one K-step of
// B fragments live and blocks LICM from hoisting weight loads.
template<int KSTEPS, int KSWITCH>
__device__ __forceinline__ void gemm_acc(v8f acc[4],
                                         const float* __restrict__ Af32,
                                         const _Float16* __restrict__ Ah,
                                         const _Float16* __restrict__ W,
                                         int mBase, int nBase0, int lane) {
#pragma unroll 1
    for (int ks = 0; ks < KSWITCH; ++ks) {
        v16h a = loadA_f(Af32, mBase, ks * 32, lane);
#pragma unroll
        for (int t = 0; t < 4; ++t) {
            v16h b = loadB_h(W, ks * 32, nBase0 + t * 16, lane);
            WMMA_F16(acc[t], a, b);
        }
    }
#pragma unroll 1
    for (int ks = KSWITCH; ks < KSTEPS; ++ks) {
        v16h a = loadA_h(Ah, mBase, (ks - KSWITCH) * 32, lane);
#pragma unroll
        for (int t = 0; t < 4; ++t) {
            v16h b = loadB_h(W, ks * 32, nBase0 + t * 16, lane);
            WMMA_F16(acc[t], a, b);
        }
    }
}

// ---------------------------------------------------------------------------
// Prep kernel: convert the 4 weight matrices f32 -> f16 into workspace.
// wh layout: mW1h[16384] | mW2h[16384] | aW1h[32768] | aW2h[16384]
// ---------------------------------------------------------------------------
__global__ void convert_weights_kernel(const float* __restrict__ mW1,
                                       const float* __restrict__ mW2,
                                       const float* __restrict__ aW1,
                                       const float* __restrict__ aW2,
                                       _Float16* __restrict__ wh) {
    const int i = blockIdx.x * blockDim.x + threadIdx.x;
    if (i < 16384)       wh[i] = (_Float16)mW1[i];
    else if (i < 32768)  wh[i] = (_Float16)mW2[i - 16384];
    else if (i < 65536)  wh[i] = (_Float16)aW1[i - 32768];
    else if (i < 81920)  wh[i] = (_Float16)aW2[i - 65536];
}

// ---------------------------------------------------------------------------
// Main fused kernel. One block = 8 batch elements (64 rows of [rows, 128]).
// ---------------------------------------------------------------------------
__global__ __launch_bounds__(THREADS)
void mp_fused_kernel(const float* __restrict__ states,     // [B, M, D]
                     const float* __restrict__ conn,       // [M, M]
                     const float* __restrict__ mweights,   // [B, M]
                     const float* __restrict__ mb1, const float* __restrict__ mb2,
                     const float* __restrict__ ab1, const float* __restrict__ ab2,
                     const _Float16* __restrict__ wh,      // packed f16 weights
                     float* __restrict__ out)              // [B, M, D]
{
    __shared__ __align__(16) float    curf[ROWS * D];   // 32 KB: f32 residual state
    __shared__ __align__(16) _Float16 bufA[ROWS * D];   // 16 KB: h, then recv
    __shared__ __align__(16) _Float16 bufB[ROWS * D];   // 16 KB: messages, then g

    const _Float16* mW1h = wh;
    const _Float16* mW2h = wh + 16384;
    const _Float16* aW1h = wh + 32768;   // [256][128]
    const _Float16* aW2h = wh + 65536;

    const int tid    = threadIdx.x;
    const int lane   = tid & 31;
    const int wave   = tid >> 5;
    const int mBase  = (wave >> 1) * 16;     // this wave's 16-row slab
    const int nBase0 = (wave & 1) * 64;      // this wave's 64-column half
    const size_t eBase = (size_t)blockIdx.x * ROWS * D;

    const int halfc = lane >> 4;             // C/D layout: M half select
    const int colc  = lane & 15;             // C/D layout: N = lane&15

    // ---- load current-state tile (f32, coalesced float4) ----
    {
        const float4* src = (const float4*)(states + eBase);
        float4* dst = (float4*)curf;
#pragma unroll
        for (int i = 0; i < (ROWS * D / 4) / THREADS; ++i)
            dst[i * THREADS + tid] = src[i * THREADS + tid];
    }
    __syncthreads();

    const v8f vz = {0.f, 0.f, 0.f, 0.f, 0.f, 0.f, 0.f, 0.f};

#pragma unroll 1
    for (int step = 0; step < NSTEPS; ++step) {
        // ===== GEMM1: h = relu(cur @ mW1 + mb1) -> bufA =====
        {
            v8f acc[4] = {vz, vz, vz, vz};
            gemm_acc<4, 4>(acc, curf, bufA /*unused*/, mW1h, mBase, nBase0, lane);
#pragma unroll
            for (int t = 0; t < 4; ++t) {
                const int col = nBase0 + t * 16 + colc;
                const float bias = mb1[col];
#pragma unroll
                for (int r = 0; r < 8; ++r) {
                    const int row = mBase + r + 8 * halfc;
                    const float v = acc[t][r] + bias;
                    bufA[row * D + col] = (_Float16)(v > 0.f ? v : 0.f);
                }
            }
        }
        __syncthreads();

        // ===== GEMM2: messages = h @ mW2 + mb2 -> bufB =====
        {
            v8f acc[4] = {vz, vz, vz, vz};
            gemm_acc<4, 0>(acc, curf /*unused*/, bufA, mW2h, mBase, nBase0, lane);
#pragma unroll
            for (int t = 0; t < 4; ++t) {
                const int col = nBase0 + t * 16 + colc;
                const float bias = mb2[col];
#pragma unroll
                for (int r = 0; r < 8; ++r) {
                    const int row = mBase + r + 8 * halfc;
                    bufB[row * D + col] = (_Float16)(acc[t][r] + bias);
                }
            }
        }
        __syncthreads();

        // ===== aggregation: recv[b,m,:] = mw[b,m] * sum_n C[m,n]*msg[b,n,:] -> bufA =====
        {
            const int row   = tid >> 2;           // 0..63
            const int dBase = (tid & 3) * 32;     // 32 d-values per thread
            const int bl    = row >> 3;
            const int m     = row & 7;
            float cm[NMOD];
#pragma unroll
            for (int n = 0; n < NMOD; ++n) cm[n] = conn[m * NMOD + n];
            const float wrow =
                mweights[(size_t)(blockIdx.x * BPB + bl) * NMOD + m];
            const _Float16* msgb = bufB + (bl * NMOD) * D;
#pragma unroll 1
            for (int d = 0; d < 32; ++d) {
                float s = 0.f;
#pragma unroll
                for (int n = 0; n < NMOD; ++n)
                    s += cm[n] * (float)msgb[n * D + dBase + d];
                bufA[row * D + dBase + d] = (_Float16)(s * wrow);
            }
        }
        __syncthreads();

        // ===== GEMM3: g = relu([cur | recv] @ aW1 + ab1) -> bufB  (K = 256) =====
        {
            v8f acc[4] = {vz, vz, vz, vz};
            gemm_acc<8, 4>(acc, curf, bufA, aW1h, mBase, nBase0, lane);
#pragma unroll
            for (int t = 0; t < 4; ++t) {
                const int col = nBase0 + t * 16 + colc;
                const float bias = ab1[col];
#pragma unroll
                for (int r = 0; r < 8; ++r) {
                    const int row = mBase + r + 8 * halfc;
                    const float v = acc[t][r] + bias;
                    bufB[row * D + col] = (_Float16)(v > 0.f ? v : 0.f);
                }
            }
        }
        __syncthreads();

        // ===== GEMM4: cur += g @ aW2 + ab2 (residual, f32 in LDS) =====
        {
            v8f acc[4] = {vz, vz, vz, vz};
            gemm_acc<4, 0>(acc, curf /*unused*/, bufB, aW2h, mBase, nBase0, lane);
#pragma unroll
            for (int t = 0; t < 4; ++t) {
                const int col = nBase0 + t * 16 + colc;
                const float bias = ab2[col];
#pragma unroll
                for (int r = 0; r < 8; ++r) {
                    const int row = mBase + r + 8 * halfc;
                    curf[row * D + col] += acc[t][r] + bias;   // exclusive per lane
                }
            }
        }
        __syncthreads();
    }

    // ---- store result tile (f32, coalesced float4) ----
    {
        const float4* src = (const float4*)curf;
        float4* dst = (float4*)(out + eBase);
#pragma unroll
        for (int i = 0; i < (ROWS * D / 4) / THREADS; ++i)
            dst[i * THREADS + tid] = src[i * THREADS + tid];
    }
}

// ---------------------------------------------------------------------------
// Host launcher.
// Input order: module_states, connection_matrix, module_weights,
//              mW1, mb1, mW2, mb2, aW1, ab1, aW2, ab2
// ---------------------------------------------------------------------------
extern "C" void kernel_launch(void* const* d_in, const int* in_sizes, int n_in,
                              void* d_out, int out_size, void* d_ws, size_t ws_size,
                              hipStream_t stream) {
    (void)in_sizes; (void)n_in; (void)out_size; (void)ws_size;

    const float* states = (const float*)d_in[0];
    const float* conn   = (const float*)d_in[1];
    const float* mw     = (const float*)d_in[2];
    const float* mW1    = (const float*)d_in[3];
    const float* mb1    = (const float*)d_in[4];
    const float* mW2    = (const float*)d_in[5];
    const float* mb2    = (const float*)d_in[6];
    const float* aW1    = (const float*)d_in[7];
    const float* ab1    = (const float*)d_in[8];
    const float* aW2    = (const float*)d_in[9];
    const float* ab2    = (const float*)d_in[10];

    _Float16* wh   = (_Float16*)d_ws;   // 81920 halves = 160 KB
    float*    outp = (float*)d_out;

    // 1) f32 -> f16 weight conversion (weights stay hot in 192 MB L2).
    convert_weights_kernel<<<(81920 + 255) / 256, 256, 0, stream>>>(
        mW1, mW2, aW1, aW2, wh);

    // 2) fused 2-step message-passing pipeline, 8 batch elements per block.
    mp_fused_kernel<<<BATCH / BPB, THREADS, 0, stream>>>(
        states, conn, mw, mb1, mb2, ab1, ab2, wh, outp);
}
